// Attention_60224031424957
// MI455X (gfx1250) — matrix operand order repaired and checked by reading, not compiled, not run on hardware
//
#include <hip/hip_runtime.h>

// ---------------- problem constants ----------------
#define B_     8
#define C_     384
#define N_     1024
#define HEADS_ 12
#define HD_    32

typedef __attribute__((ext_vector_type(16))) __bf16       v16bf;
typedef __attribute__((ext_vector_type(8)))  float        v8f;
typedef __attribute__((ext_vector_type(4)))  unsigned int u32x4;
typedef __attribute__((ext_vector_type(4)))  float        f32x4;
typedef __attribute__((ext_vector_type(4)))  int          i32x4;
typedef __attribute__((ext_vector_type(8)))  int          i32x8;

union FragBF { v16bf v; u32x4 q[2]; };

// native bf16 convert (lets the compiler emit v_cvt_*_bf16_f32)
__device__ __forceinline__ unsigned short f2bf(float f) {
    __bf16 h = (__bf16)f;
    return __builtin_bit_cast(unsigned short, h);
}
__device__ __forceinline__ unsigned pack2(float a, float b) {
    return (unsigned)f2bf(a) | ((unsigned)f2bf(b) << 16);
}

__device__ __forceinline__ v8f wmma_bf16(v16bf a, v16bf b, v8f c) {
    // D = A(16x32) * B(32x16) + C, f32 accumulate
    return __builtin_amdgcn_wmma_f32_16x16x32_bf16(false, a, false, b,
                                                   (short)0, c, false, false);
}

// ---------------------------------------------------------------------------
// Tensor Data Mover: 2D tile (tile_d0 elems/row x tile_d1 rows, 2-byte elems)
// global -> LDS.  Descriptor per CDNA5 ISA:
//   group0: count=1 | lds_addr | global_addr[56:0] | type=2
//   group1: data_size=1(2B), tensor_dim0=row stride, tensor_dim1 big,
//           tile_dim0/1, tensor_dim0_stride (48b), rest 0 (no cluster mask)
// Issued by one wave; tracked with TENSORcnt.
// ---------------------------------------------------------------------------
__device__ __forceinline__ void tdm_load_tile_2d(
    const unsigned short* gptr, unsigned lds_off,
    unsigned tile_d0, unsigned tile_d1, unsigned long long row_stride_elems)
{
    unsigned long long ga = (unsigned long long)(uintptr_t)gptr;
    unsigned ga_lo = __builtin_amdgcn_readfirstlane((unsigned)ga);
    unsigned ga_hi = __builtin_amdgcn_readfirstlane((unsigned)(ga >> 32));
    lds_off        = __builtin_amdgcn_readfirstlane(lds_off);

    i32x4 g0;
    g0.x = 1;                                              // count=1 (user D#)
    g0.y = (int)lds_off;                                   // lds_addr
    g0.z = (int)ga_lo;                                     // global_addr[31:0]
    g0.w = (int)((ga_hi & 0x01FFFFFFu) | 0x80000000u);     // addr[56:32]|type=2

    const unsigned td0  = (unsigned)row_stride_elems;      // tensor_dim0
    const unsigned td1  = 1u << 20;                        // tensor_dim1 (big)
    const unsigned s0lo = (unsigned)row_stride_elems;
    const unsigned s0hi = (unsigned)(row_stride_elems >> 32) & 0xFFFFu;

    i32x8 g1;
    g1[0] = (int)(1u << 16);                               // data_size=1 (2B)
    g1[1] = (int)((td0 & 0xFFFFu) << 16);                  // tensor_dim0[15:0]
    g1[2] = (int)(((td0 >> 16) & 0xFFFFu) | ((td1 & 0xFFFFu) << 16));
    g1[3] = (int)(((td1 >> 16) & 0xFFFFu) | (tile_d0 << 16)); // tile_dim0
    g1[4] = (int)(tile_d1 & 0xFFFFu);                      // tile_dim1, tile_dim2=0
    g1[5] = (int)s0lo;                                     // dim0 stride lo32
    g1[6] = (int)s0hi;                                     // dim0 stride hi16
    g1[7] = 0;
    asm volatile("tensor_load_to_lds %0, %1" :: "s"(g0), "s"(g1) : "memory");
}

// ---------------------------------------------------------------------------
// Kernel 1: expand relative-position bias -> biasW[h][n][m] (f32, 48 MB;
// resident in the 192MB L2, so attention re-reads are L2 hits)
// ---------------------------------------------------------------------------
__global__ void __launch_bounds__(256)
bias_expand(const float* __restrict__ table, const int* __restrict__ rel,
            float* __restrict__ biasW)
{
    int id = blockIdx.x * 256 + threadIdx.x;   // over HEADS_*N_*N_
    int h  = id >> 20;                         // N_*N_ == 2^20
    int r  = id & 0xFFFFF;
    biasW[id] = table[rel[r] * HEADS_ + h];
}

// ---------------------------------------------------------------------------
// Kernel 2: fused QKV projection.  Stacked GEMM W[1152,384] x X[384,1024] per
// batch; 64x64 workgroup tile, 4 waves, K-steps of 32, bf16 WMMA.
// Q written pre-scaled, layouts:  Q,K: [b][head][n][d]   V: [b][head][d][n]
// ---------------------------------------------------------------------------
__global__ void __launch_bounds__(128)
qkv_proj(const float* __restrict__ x,    const float* __restrict__ q_w,
         const float* __restrict__ q_b,  const float* __restrict__ kv_w,
         const float* __restrict__ kv_b,
         unsigned short* __restrict__ Qb, unsigned short* __restrict__ Kb,
         unsigned short* __restrict__ Vb)
{
    __shared__ unsigned short Asub[64 * 32];   // weights [m][k] bf16
    __shared__ unsigned short Bsub[64 * 32];   // x tile  [n][k] bf16 (transposed)

    const int t    = threadIdx.x;
    const int wave = t >> 5, lane = t & 31;
    const int Nbase = blockIdx.x * 64;
    const int Mbase = blockIdx.y * 64;
    const int b     = blockIdx.z;
    const float scale = 0.17677669529663687f;  // 32^-0.5

    v8f acc[4];
    for (int i = 0; i < 4; ++i)
        for (int j = 0; j < 8; ++j) acc[i][j] = 0.f;

    for (int k0 = 0; k0 < C_; k0 += 32) {
        {   // stage A (weights) f32 -> bf16, packed dword LDS stores
            int r = t >> 1, kk = (t & 1) * 16;
            int ch = Mbase + r;
            const float* wrow = (ch < C_) ? (q_w + (size_t)ch * C_)
                                          : (kv_w + (size_t)(ch - C_) * C_);
            const f32x4* wv = (const f32x4*)(wrow + k0 + kk);
            f32x4 f0 = wv[0], f1 = wv[1], f2 = wv[2], f3 = wv[3];
            unsigned* dst = (unsigned*)&Asub[r * 32 + kk];
            dst[0] = pack2(f0.x, f0.y); dst[1] = pack2(f0.z, f0.w);
            dst[2] = pack2(f1.x, f1.y); dst[3] = pack2(f1.z, f1.w);
            dst[4] = pack2(f2.x, f2.y); dst[5] = pack2(f2.z, f2.w);
            dst[6] = pack2(f3.x, f3.y); dst[7] = pack2(f3.z, f3.w);
        }
        {   // stage X transposed to [n][k]
            int kr = t >> 2, nn0 = (t & 3) * 16;
            const f32x4* xv =
                (const f32x4*)(x + (size_t)(b * C_ + k0 + kr) * N_ + Nbase + nn0);
            float vals[16];
            *(f32x4*)&vals[0]  = xv[0];
            *(f32x4*)&vals[4]  = xv[1];
            *(f32x4*)&vals[8]  = xv[2];
            *(f32x4*)&vals[12] = xv[3];
            for (int j = 0; j < 16; ++j)
                Bsub[(nn0 + j) * 32 + kr] = f2bf(vals[j]);
        }
        __syncthreads();

        FragBF fa;
        {   // A fragment: per-lane row, 16 contiguous K halves
            int row = wave * 16 + (lane & 15);
            int kk  = (lane & 16) ? 16 : 0;
            fa.q[0] = *(const u32x4*)&Asub[row * 32 + kk];
            fa.q[1] = *(const u32x4*)&Asub[row * 32 + kk + 8];
        }
        for (int tt = 0; tt < 4; ++tt) {
            // B fragment: per-lane column, 16 contiguous K halves
            int col = tt * 16 + (lane & 15);
            int d0  = (lane & 16) ? 16 : 0;
            v16bf fb = *(const v16bf*)&Bsub[col * 32 + d0];
            acc[tt] = wmma_bf16(fa.v, fb, acc[tt]);
        }
        __syncthreads();
    }

    // epilogue: bias, scale(Q), scatter into Q/K/V bf16
    const int hi8 = (lane & 16) ? 8 : 0;
    for (int tt = 0; tt < 4; ++tt) {
        int n = Nbase + tt * 16 + (lane & 15);
        for (int i = 0; i < 8; ++i) {
            int   ch = Mbase + wave * 16 + i + hi8;
            float v  = acc[tt][i];
            if (ch < C_) {                                    // Q [bh][n][d]
                v = (v + q_b[ch]) * scale;
                int head = ch >> 5, d = ch & 31;
                Qb[(((size_t)b * HEADS_ + head) * N_ + n) * HD_ + d] = f2bf(v);
            } else {
                int c2 = ch - C_;
                v += kv_b[c2];
                if (c2 < C_) {                                // K [bh][n][d]
                    int head = c2 >> 5, d = c2 & 31;
                    Kb[(((size_t)b * HEADS_ + head) * N_ + n) * HD_ + d] = f2bf(v);
                } else {                                      // V [bh][d][n]
                    int c3 = c2 - C_;
                    int head = c3 >> 5, d = c3 & 31;
                    Vb[(((size_t)b * HEADS_ + head) * HD_ + d) * N_ + n] = f2bf(v);
                }
            }
        }
    }
}

// ---------------------------------------------------------------------------
// Kernel 3: flash attention.  Workgroup = (b, head, 64 query rows), 4 waves x
// 16 rows.  K/V tiles (32 keys) staged HBM->LDS by the Tensor Data Mover,
// double-buffered: tile mt+1 is in flight while tile mt runs through
// 4x v_wmma_f32_16x16x32_bf16 + online softmax.  hd==32 -> QK^T is 1 WMMA.
// ---------------------------------------------------------------------------
__global__ void __launch_bounds__(128)
flash_attn(const unsigned short* __restrict__ Qb,
           const unsigned short* __restrict__ Kb,
           const unsigned short* __restrict__ Vb,
           const float*          __restrict__ biasW,
           unsigned short*       __restrict__ Ob)
{
    __shared__ unsigned short Kt[2][32 * 32];     // K tile [m][d], double buffer
    __shared__ unsigned short Vt[2][32 * 32];     // V tile [d][m], double buffer
    __shared__ unsigned short Plds[4][16 * 32];   // per-wave P bounce (acc->Afrag)

    const int t    = threadIdx.x;
    const int wave = t >> 5, lane = t & 31;
    const int h    = blockIdx.y;
    const int b    = blockIdx.z;
    const size_t bh = (size_t)b * HEADS_ + h;
    const int nbase = blockIdx.x * 64 + wave * 16;
    const int hi8   = (lane & 16) ? 8 : 0;
    const int lc    = lane & 15;
    const int NT    = N_ / 32;

    const unsigned kt_lds = (unsigned)(uintptr_t)&Kt[0][0];
    const unsigned vt_lds = (unsigned)(uintptr_t)&Vt[0][0];
    const unsigned short* Kbase = Kb + bh * N_ * HD_;   // rows [m][d]
    const unsigned short* Vbase = Vb + bh * HD_ * N_;   // rows [d][n]

    // ---- Q fragment (held for the whole kernel; already scaled) -----------
    FragBF qf;
    {
        const unsigned short* qrow = Qb + (bh * N_ + (nbase + lc)) * HD_;
        int kk = (lane & 16) ? 16 : 0;
        qf.q[0] = *(const u32x4*)(qrow + kk);
        qf.q[1] = *(const u32x4*)(qrow + kk + 8);
    }

    float m_run[8], l_run[8];
    v8f O0, O1;
    for (int i = 0; i < 8; ++i) { m_run[i] = -3.0e38f; l_run[i] = 0.f; O0[i] = 0.f; O1[i] = 0.f; }

    if (wave == 0) {                                   // preload tile 0
        tdm_load_tile_2d(Kbase, kt_lds, 32, 32, HD_);
        tdm_load_tile_2d(Vbase, vt_lds, 32, 32, N_);
    }

    for (int mt = 0; mt < NT; ++mt) {
        const int m0  = mt * 32;
        const int buf = mt & 1;
        if (wave == 0) {
            if (mt + 1 < NT) {                         // async-issue next tile
                tdm_load_tile_2d(Kbase + (size_t)(m0 + 32) * HD_,
                                 kt_lds + (buf ^ 1) * 2048, 32, 32, HD_);
                tdm_load_tile_2d(Vbase + (m0 + 32),
                                 vt_lds + (buf ^ 1) * 2048, 32, 32, N_);
                __builtin_amdgcn_s_wait_tensorcnt(2);  // drain current tile only
            } else {
                __builtin_amdgcn_s_wait_tensorcnt(0);
            }
        }
        __syncthreads();

        // ---- S = Q K^T (two 16-col tiles; hd==32 -> one WMMA each) --------
        v8f s0, s1;
        for (int i = 0; i < 8; ++i) { s0[i] = 0.f; s1[i] = 0.f; }
        {
            int d0 = (lane & 16) ? 16 : 0;
            v16bf kf0 = *(const v16bf*)&Kt[buf][lc * 32 + d0];
            v16bf kf1 = *(const v16bf*)&Kt[buf][(lc + 16) * 32 + d0];
            s0 = wmma_bf16(qf.v, kf0, s0);
            s1 = wmma_bf16(qf.v, kf1, s1);
        }

        // ---- bias + online softmax (row stats across 16-lane groups) ------
        for (int i = 0; i < 8; ++i) {
            int n_i = nbase + i + hi8;
            const float* brow = biasW + ((size_t)h * N_ + n_i) * N_ + m0;
            float a0 = s0[i] + brow[lc];
            float a1 = s1[i] + brow[16 + lc];
            float mx = fmaxf(a0, a1);
            for (int s = 1; s < 16; s <<= 1) mx = fmaxf(mx, __shfl_xor(mx, s, 32));
            float m_new = fmaxf(m_run[i], mx);
            float alpha = __expf(m_run[i] - m_new);
            float p0 = __expf(a0 - m_new);
            float p1 = __expf(a1 - m_new);
            float rs = p0 + p1;
            for (int s = 1; s < 16; s <<= 1) rs += __shfl_xor(rs, s, 32);
            l_run[i] = l_run[i] * alpha + rs;
            m_run[i] = m_new;
            O0[i] *= alpha;
            O1[i] *= alpha;
            int r = i + hi8;                 // acc layout -> A layout via LDS
            Plds[wave][r * 32 + lc]      = f2bf(p0);
            Plds[wave][r * 32 + 16 + lc] = f2bf(p1);
        }

        // ---- O += P V  (wave-private LDS bounce: DS in-order, no barrier) -
        {
            FragBF pf;
            int kk = (lane & 16) ? 16 : 0;
            pf.q[0] = *(const u32x4*)&Plds[wave][lc * 32 + kk];
            pf.q[1] = *(const u32x4*)&Plds[wave][lc * 32 + kk + 8];
            int d0 = (lane & 16) ? 16 : 0;
            v16bf vf0 = *(const v16bf*)&Vt[buf][lc * 32 + d0];
            v16bf vf1 = *(const v16bf*)&Vt[buf][(lc + 16) * 32 + d0];
            O0 = wmma_bf16(pf.v, vf0, O0);
            O1 = wmma_bf16(pf.v, vf1, O1);
        }
        __syncthreads();     // all waves done with buf before TDM overwrites it
    }

    // ---- normalize, store in proj-input layout [b][c=h*32+d][n] -----------
    for (int i = 0; i < 8; ++i) {
        float inv = 1.f / l_run[i];
        int n_i = nbase + i + hi8;
        int c0 = h * HD_ + lc;
        int c1 = h * HD_ + 16 + lc;
        Ob[((size_t)b * C_ + c0) * N_ + n_i] = f2bf(O0[i] * inv);
        Ob[((size_t)b * C_ + c1) * N_ + n_i] = f2bf(O1[i] * inv);
    }
}

// ---------------------------------------------------------------------------
// Kernel 4: output projection.  proj_w[384,384] x Ob[384,1024] + proj_b -> f32
// ---------------------------------------------------------------------------
__global__ void __launch_bounds__(128)
proj_gemm(const unsigned short* __restrict__ Ob, const float* __restrict__ proj_w,
          const float* __restrict__ proj_b, float* __restrict__ out)
{
    __shared__ unsigned short Asub[64 * 32];
    __shared__ unsigned short Bsub[64 * 32];

    const int t    = threadIdx.x;
    const int wave = t >> 5, lane = t & 31;
    const int Nbase = blockIdx.x * 64;
    const int Mbase = blockIdx.y * 64;
    const int b     = blockIdx.z;

    v8f acc[4];
    for (int i = 0; i < 4; ++i)
        for (int j = 0; j < 8; ++j) acc[i][j] = 0.f;

    for (int k0 = 0; k0 < C_; k0 += 32) {
        {
            int r = t >> 1, kk = (t & 1) * 16;
            const f32x4* wv =
                (const f32x4*)(proj_w + (size_t)(Mbase + r) * C_ + k0 + kk);
            f32x4 f0 = wv[0], f1 = wv[1], f2 = wv[2], f3 = wv[3];
            unsigned* dst = (unsigned*)&Asub[r * 32 + kk];
            dst[0] = pack2(f0.x, f0.y); dst[1] = pack2(f0.z, f0.w);
            dst[2] = pack2(f1.x, f1.y); dst[3] = pack2(f1.z, f1.w);
            dst[4] = pack2(f2.x, f2.y); dst[5] = pack2(f2.z, f2.w);
            dst[6] = pack2(f3.x, f3.y); dst[7] = pack2(f3.z, f3.w);
        }
        {
            int kr = t >> 2, nn0 = (t & 3) * 16;
            const u32x4* ov = (const u32x4*)
                (Ob + ((size_t)b * C_ + k0 + kr) * N_ + Nbase + nn0);
            unsigned short s16[16];
            *(u32x4*)&s16[0] = ov[0];
            *(u32x4*)&s16[8] = ov[1];
            for (int j = 0; j < 16; ++j)
                Bsub[(nn0 + j) * 32 + kr] = s16[j];
        }
        __syncthreads();

        FragBF fa;
        {
            int row = wave * 16 + (lane & 15);
            int kk  = (lane & 16) ? 16 : 0;
            fa.q[0] = *(const u32x4*)&Asub[row * 32 + kk];
            fa.q[1] = *(const u32x4*)&Asub[row * 32 + kk + 8];
        }
        for (int tt = 0; tt < 4; ++tt) {
            int col = tt * 16 + (lane & 15);
            int d0  = (lane & 16) ? 16 : 0;
            v16bf fb = *(const v16bf*)&Bsub[col * 32 + d0];
            acc[tt] = wmma_bf16(fa.v, fb, acc[tt]);
        }
        __syncthreads();
    }

    const int hi8 = (lane & 16) ? 8 : 0;
    for (int tt = 0; tt < 4; ++tt) {
        int n = Nbase + tt * 16 + (lane & 15);
        for (int i = 0; i < 8; ++i) {
            int ch = Mbase + wave * 16 + i + hi8;
            out[((size_t)b * C_ + ch) * N_ + n] = acc[tt][i] + proj_b[ch];
        }
    }
}

// ---------------------------------------------------------------------------
// Host launcher
// ---------------------------------------------------------------------------
extern "C" void kernel_launch(void* const* d_in, const int* in_sizes, int n_in,
                              void* d_out, int out_size, void* d_ws, size_t ws_size,
                              hipStream_t stream) {
    (void)in_sizes; (void)n_in; (void)out_size; (void)ws_size;
    const float* x      = (const float*)d_in[0];
    const float* q_w    = (const float*)d_in[1];
    const float* q_b    = (const float*)d_in[2];
    const float* kv_w   = (const float*)d_in[3];
    const float* kv_b   = (const float*)d_in[4];
    const float* proj_w = (const float*)d_in[5];
    const float* proj_b = (const float*)d_in[6];
    const float* rpb    = (const float*)d_in[7];
    const int*   rel    = (const int*)d_in[8];

    // workspace: Qb | Kb | Vb | Ob (bf16) | biasW (f32)  ~= 75.5 MB
    char* ws = (char*)d_ws;
    const size_t szQKV = (size_t)B_ * HEADS_ * N_ * HD_ * sizeof(unsigned short);
    unsigned short* Qb = (unsigned short*)(ws);
    unsigned short* Kb = (unsigned short*)(ws + 1 * szQKV);
    unsigned short* Vb = (unsigned short*)(ws + 2 * szQKV);
    unsigned short* Ob = (unsigned short*)(ws + 3 * szQKV);
    float*          biasW = (float*)(ws + 4 * szQKV);

    bias_expand<<<dim3((HEADS_ * N_ * N_) / 256), dim3(256), 0, stream>>>(rpb, rel, biasW);
    qkv_proj  <<<dim3(N_ / 64, (3 * C_) / 64, B_), dim3(128), 0, stream>>>(
        x, q_w, q_b, kv_w, kv_b, Qb, Kb, Vb);
    flash_attn<<<dim3(N_ / 64, HEADS_, B_), dim3(128), 0, stream>>>(
        Qb, Kb, Vb, biasW, Ob);
    proj_gemm <<<dim3(N_ / 64, C_ / 64, B_), dim3(128), 0, stream>>>(
        Ob, proj_w, proj_b, (float*)d_out);
}
